// MultiHeadAttentionClassical_65481071404409
// MI455X (gfx1250) — compile-verified, matrix operand order checked by reading, not run on hardware
//
#include <hip/hip_runtime.h>
#include <hip/hip_bf16.h>

#define S_LEN 4096
#define E_DIM 768
#define H_NUM 12
#define D_HEAD 64

typedef __attribute__((ext_vector_type(16))) _Float16 v16h;
typedef __attribute__((ext_vector_type(8)))  _Float16 v8h;
typedef __attribute__((ext_vector_type(8)))  float    v8f;
typedef __attribute__((ext_vector_type(4)))  unsigned int v4u;
typedef __attribute__((ext_vector_type(4)))  int      v4i;
typedef __attribute__((ext_vector_type(8)))  int      v8i;

#if defined(__has_builtin)
#if __has_builtin(__builtin_amdgcn_tensor_load_to_lds) && \
    __has_builtin(__builtin_amdgcn_s_wait_tensorcnt)
#define HAVE_TDM 1
#endif
#endif

union frag_u { v16h v; struct { v8h lo, hi; } s; };

// Load a 16x32 (A) or 32x16 (B) f16 fragment from row-major memory where the
// WMMA K-dimension is contiguous. Per ISA layout: lanes 0-15 hold k={0..7,16..23},
// lanes 16-31 hold k={8..15,24..31}; caller supplies the row (M for A, N for B).
// Works for both global and LDS pointers (address-space inferred after inlining).
__device__ __forceinline__ v16h load_frag_k32(const _Float16* base,
                                              int row, int stride, int kbase, int lane) {
  const _Float16* p = base + (size_t)row * stride + kbase + ((lane & 16) ? 8 : 0);
  frag_u u;
  u.s.lo = *(const v8h*)(p);
  u.s.hi = *(const v8h*)(p + 16);
  return u.v;
}

__device__ __forceinline__ v8f wmma_f16(v16h a, v16h b, v8f c) {
  return __builtin_amdgcn_wmma_f32_16x16x32_f16(false, a, false, b, (short)0, c,
                                                false, false);
}

// ---------------------------------------------------------------------------
// TDM: 2-D f16 tile load (tile_d1 rows x tile_d0 elements) from a row-major
// tensor (tensor_d1 x tensor_d0, row stride stride0 elements) into LDS,
// packed contiguously. Descriptor per CDNA5 ISA 8.3/8.4. Executed by one wave.
// ---------------------------------------------------------------------------
__device__ __forceinline__ void tdm_load_2d(_Float16* lds_ptr, const _Float16* gaddr,
                                            unsigned tensor_d0, unsigned tensor_d1,
                                            unsigned tile_d0, unsigned tile_d1,
                                            unsigned stride0, int lane) {
#ifdef HAVE_TDM
  unsigned long long ga = (unsigned long long)(uintptr_t)gaddr;
  unsigned lds_off = (unsigned)(uintptr_t)lds_ptr;  // low 32 bits = LDS byte offset
  v4u g0;
  g0.x = 1u;                                        // count=1 (valid), user mode
  g0.y = lds_off;                                   // lds_addr
  g0.z = (unsigned)(ga & 0xffffffffu);              // global_addr[31:0]
  g0.w = (unsigned)((ga >> 32) & 0x1ffffffu) | (2u << 30);  // ga[56:32] | type=2
  v8i g1;
  g1[0] = (int)(1u << 16);                          // data_size=1 -> 2 bytes
  g1[1] = (int)((tensor_d0 & 0xffffu) << 16);       // tensor_dim0[15:0]
  g1[2] = (int)((tensor_d0 >> 16) | ((tensor_d1 & 0xffffu) << 16));
  g1[3] = (int)((tensor_d1 >> 16) | (tile_d0 << 16));
  g1[4] = (int)tile_d1;                             // tile_dim1 (tile_dim2=0)
  g1[5] = (int)stride0;                             // tensor_dim0_stride[31:0]
  g1[6] = (int)(stride0 >> 16 >> 16);               // stride[47:32], dim1_stride=0
  g1[7] = 0;
  v4i z4 = {0, 0, 0, 0};
#if __clang_major__ >= 23
  v8i z8 = {0, 0, 0, 0, 0, 0, 0, 0};
  __builtin_amdgcn_tensor_load_to_lds(g0, g1, z4, z4, z8, 0);
#else
  __builtin_amdgcn_tensor_load_to_lds(g0, g1, z4, z4, 0);
#endif
#else
  // Fallback: cooperative copy by the issuing wave (v8h chunks).
  (void)tensor_d0; (void)tensor_d1;
  const unsigned total = tile_d0 * tile_d1;
  for (unsigned idx = (unsigned)lane * 8; idx < total; idx += 32u * 8u) {
    unsigned row = idx / tile_d0, col = idx % tile_d0;
    *(v8h*)(lds_ptr + idx) = *(const v8h*)(gaddr + (size_t)row * stride0 + col);
  }
#endif
}

__device__ __forceinline__ void tdm_wait2() {
#ifdef HAVE_TDM
  __builtin_amdgcn_s_wait_tensorcnt(2);
#endif
}
__device__ __forceinline__ void tdm_wait0() {
#ifdef HAVE_TDM
  __builtin_amdgcn_s_wait_tensorcnt(0);
#endif
}

__global__ void cvt_f32_f16_kernel(const float* __restrict__ in,
                                   _Float16* __restrict__ out, int n) {
  int i = blockIdx.x * blockDim.x + threadIdx.x;
  if (i < n) out[i] = (_Float16)in[i];
}

// Fused QKV projection: C = x * W^T + b for W in {Wq,Wk,Wv}.
// Each wave computes a 16(M) x 64(N) strip; N-strip == exactly one head.
// Q stored [h][s][d] pre-scaled by 1/sqrt(D); K stored [h][s][d]; V stored
// transposed [h][d][s] so attention's P*V B-fragments are contiguous in key.
__global__ void qkv_gemm_kernel(const _Float16* __restrict__ xh,
                                const _Float16* __restrict__ Wqh,
                                const _Float16* __restrict__ Wkh,
                                const _Float16* __restrict__ Wvh,
                                const float* __restrict__ bq,
                                const float* __restrict__ bk,
                                const float* __restrict__ bv,
                                _Float16* __restrict__ Qh,
                                _Float16* __restrict__ Kh,
                                _Float16* __restrict__ Vt) {
  const int lane = threadIdx.x & 31;
  const int w = blockIdx.x * (blockDim.x >> 5) + (threadIdx.x >> 5);
  const int MT = S_LEN / 16, NS = E_DIM / 64;
  if (w >= 3 * MT * NS) return;
  const int mat = w / (MT * NS);
  const int rem = w % (MT * NS);
  const int m0 = (rem / NS) * 16;
  const int h  = rem % NS;
  const int n0 = h * 64;
  const _Float16* W    = (mat == 0) ? Wqh : (mat == 1) ? Wkh : Wvh;
  const float*    bias = (mat == 0) ? bq  : (mat == 1) ? bk  : bv;

  v8f acc[4] = {v8f{}, v8f{}, v8f{}, v8f{}};
  const int arow = m0 + (lane & 15);
  for (int k0 = 0; k0 < E_DIM; k0 += 32) {
    v16h a = load_frag_k32(xh, arow, E_DIM, k0, lane);
#pragma unroll
    for (int t = 0; t < 4; ++t) {
      v16h b = load_frag_k32(W, n0 + t * 16 + (lane & 15), E_DIM, k0, lane);
      acc[t] = wmma_f16(a, b, acc[t]);
    }
  }

  const int mrow = m0 + 8 * (lane >> 4);
#pragma unroll
  for (int t = 0; t < 4; ++t) {
#pragma unroll
    for (int r = 0; r < 8; ++r) {
      const int s = mrow + r;
      const int d = t * 16 + (lane & 15);
      float val = acc[t][r] + bias[n0 + d];
      if (mat == 0) {
        Qh[((size_t)h * S_LEN + s) * D_HEAD + d] = (_Float16)(val * 0.125f);
      } else if (mat == 1) {
        Kh[((size_t)h * S_LEN + s) * D_HEAD + d] = (_Float16)val;
      } else {
        Vt[((size_t)h * D_HEAD + d) * S_LEN + s] = (_Float16)val;
      }
    }
  }
}

// Flash-attention. Block = 8 waves, all in the SAME head, 8 consecutive
// 16-query tiles. K/V 32-key blocks are staged once per workgroup into LDS by
// wave 0 via the Tensor Data Mover (double-buffered, s_wait_tensorcnt), then
// all 8 waves consume the staged tiles via ds_load_b128 fragments.
// Scores computed transposed (S^T = K*Q^T) so the f32 C layout maps
// lane-locally onto the f16 A layout of P for the P*V WMMA.
__global__ void attn_kernel(const _Float16* __restrict__ Qh,
                            const _Float16* __restrict__ Kh,
                            const _Float16* __restrict__ Vt,
                            _Float16* __restrict__ Ah) {
  __shared__ _Float16 skbuf[2][32 * D_HEAD];   // K block: 32 keys x 64 d
  __shared__ _Float16 svbuf[2][D_HEAD * 32];   // V block: 64 d rows x 32 keys

  const int lane = threadIdx.x & 31;
  const int wid  = threadIdx.x >> 5;
  const int h     = blockIdx.x >> 5;           // 12 heads * 32 blocks
  const int qbase = (blockIdx.x & 31) * 128;   // 8 waves * 16 queries
  const int q0    = qbase + wid * 16;

  const _Float16* Qp = Qh + (size_t)h * S_LEN * D_HEAD;
  const _Float16* Kp = Kh + (size_t)h * S_LEN * D_HEAD;
  const _Float16* Vp = Vt + (size_t)h * D_HEAD * S_LEN;

  const v16h bq0 = load_frag_k32(Qp, q0 + (lane & 15), D_HEAD, 0,  lane);
  const v16h bq1 = load_frag_k32(Qp, q0 + (lane & 15), D_HEAD, 32, lane);

  float mprev = -1e30f, lsum = 0.f;
  v8f o[4] = {v8f{}, v8f{}, v8f{}, v8f{}};
  const int hi8 = (lane & 16) ? 8 : 0;

  const int NB = S_LEN / 32;                   // 128 key blocks
  if (wid == 0) {
    tdm_load_2d(skbuf[0], Kp, D_HEAD, S_LEN, D_HEAD, 32, D_HEAD, lane);
    tdm_load_2d(svbuf[0], Vp, S_LEN, D_HEAD, 32, D_HEAD, S_LEN, lane);
  }

  for (int ib = 0; ib < NB; ++ib) {
    const int cur = ib & 1;
    if (wid == 0) {
      if (ib + 1 < NB) {
        const int kb1 = (ib + 1) * 32;
        tdm_load_2d(skbuf[cur ^ 1], Kp + (size_t)kb1 * D_HEAD,
                    D_HEAD, S_LEN, D_HEAD, 32, D_HEAD, lane);
        tdm_load_2d(svbuf[cur ^ 1], Vp + kb1,
                    S_LEN, D_HEAD, 32, D_HEAD, S_LEN, lane);
        tdm_wait2();                           // current pair done; next in flight
      } else {
        tdm_wait0();
      }
    }
    __syncthreads();                           // staged tiles visible to all waves

    const _Float16* kl = skbuf[cur];
    const _Float16* vl = svbuf[cur];

    v16h a00 = load_frag_k32(kl,      (lane & 15), D_HEAD, 0,  lane);
    v16h a01 = load_frag_k32(kl,      (lane & 15), D_HEAD, 32, lane);
    v16h a10 = load_frag_k32(kl, 16 + (lane & 15), D_HEAD, 0,  lane);
    v16h a11 = load_frag_k32(kl, 16 + (lane & 15), D_HEAD, 32, lane);

    v8f z = {};
    v8f ct0 = wmma_f16(a00, bq0, z);  ct0 = wmma_f16(a01, bq1, ct0);
    v8f ct1 = wmma_f16(a10, bq0, z);  ct1 = wmma_f16(a11, bq1, ct1);

    // Row (=query) max: lane-local over 16 keys + partner lane via xor-16.
    float tmax = -1e30f;
#pragma unroll
    for (int r = 0; r < 8; ++r) tmax = fmaxf(tmax, fmaxf(ct0[r], ct1[r]));
    tmax = fmaxf(tmax, __shfl_xor(tmax, 16, 32));
    const float mnew  = fmaxf(mprev, tmax);
    const float alpha = __expf(mprev - mnew);

    frag_u pu;
    float rsum = 0.f;
#pragma unroll
    for (int r = 0; r < 8; ++r) {
      float e0 = __expf(ct0[r] - mnew);
      float e1 = __expf(ct1[r] - mnew);
      rsum += e0 + e1;
      pu.s.lo[r] = (_Float16)e0;   // keys {0..7}|{8..15}   -> A halves 0..7
      pu.s.hi[r] = (_Float16)e1;   // keys {16..23}|{24..31} -> A halves 8..15
    }
    rsum += __shfl_xor(rsum, 16, 32);
    lsum  = lsum * alpha + rsum;
    mprev = mnew;

    // Rescale accumulator: per-output-row factor broadcast from owning lane.
    float fr[8];
#pragma unroll
    for (int r = 0; r < 8; ++r) fr[r] = __shfl(alpha, r + hi8, 32);
#pragma unroll
    for (int t = 0; t < 4; ++t)
#pragma unroll
      for (int r = 0; r < 8; ++r) o[t][r] *= fr[r];

    const v16h pa = pu.v;
#pragma unroll
    for (int t = 0; t < 4; ++t) {
      v16h bv = load_frag_k32(vl, t * 16 + (lane & 15), 32, 0, lane);
      o[t] = wmma_f16(pa, bv, o[t]);
    }
    __syncthreads();                           // all reads done before overwrite
  }

  const float rcp = 1.f / lsum;
  float fr[8];
#pragma unroll
  for (int r = 0; r < 8; ++r) fr[r] = __shfl(rcp, r + hi8, 32);
#pragma unroll
  for (int t = 0; t < 4; ++t)
#pragma unroll
    for (int r = 0; r < 8; ++r) {
      const int s = q0 + r + 8 * (lane >> 4);
      const int e = h * D_HEAD + t * 16 + (lane & 15);
      Ah[(size_t)s * E_DIM + e] = (_Float16)(o[t][r] * fr[r]);
    }
}

// Output projection: out = Ah * Wo^T + bo (fp32 output).
__global__ void oproj_kernel(const _Float16* __restrict__ Ah,
                             const _Float16* __restrict__ Woh,
                             const float* __restrict__ bo,
                             float* __restrict__ out) {
  const int lane = threadIdx.x & 31;
  const int w = blockIdx.x * (blockDim.x >> 5) + (threadIdx.x >> 5);
  const int MT = S_LEN / 16, NS = E_DIM / 64;
  if (w >= MT * NS) return;
  const int m0 = (w / NS) * 16;
  const int n0 = (w % NS) * 64;

  v8f acc[4] = {v8f{}, v8f{}, v8f{}, v8f{}};
  const int arow = m0 + (lane & 15);
  for (int k0 = 0; k0 < E_DIM; k0 += 32) {
    v16h a = load_frag_k32(Ah, arow, E_DIM, k0, lane);
#pragma unroll
    for (int t = 0; t < 4; ++t) {
      v16h b = load_frag_k32(Woh, n0 + t * 16 + (lane & 15), E_DIM, k0, lane);
      acc[t] = wmma_f16(a, b, acc[t]);
    }
  }

  const int mrow = m0 + 8 * (lane >> 4);
#pragma unroll
  for (int t = 0; t < 4; ++t)
#pragma unroll
    for (int r = 0; r < 8; ++r) {
      const int n = n0 + t * 16 + (lane & 15);
      out[(size_t)(mrow + r) * E_DIM + n] = acc[t][r] + bo[n];
    }
}

extern "C" void kernel_launch(void* const* d_in, const int* in_sizes, int n_in,
                              void* d_out, int out_size, void* d_ws, size_t ws_size,
                              hipStream_t stream) {
  (void)in_sizes; (void)n_in; (void)out_size; (void)ws_size;
  const float* x  = (const float*)d_in[0];
  const float* Wq = (const float*)d_in[1];
  const float* bq = (const float*)d_in[2];
  const float* Wk = (const float*)d_in[3];
  const float* bk = (const float*)d_in[4];
  const float* Wv = (const float*)d_in[5];
  const float* bv = (const float*)d_in[6];
  const float* Wo = (const float*)d_in[7];
  const float* bo = (const float*)d_in[8];
  float* out = (float*)d_out;

  const size_t NX = (size_t)S_LEN * E_DIM;   // 3,145,728
  const size_t NW = (size_t)E_DIM * E_DIM;   //   589,824
  char* ws = (char*)d_ws;
  size_t off = 0;
  _Float16* xh  = (_Float16*)(ws + off); off += NX * 2;
  _Float16* Wqh = (_Float16*)(ws + off); off += NW * 2;
  _Float16* Wkh = (_Float16*)(ws + off); off += NW * 2;
  _Float16* Wvh = (_Float16*)(ws + off); off += NW * 2;
  _Float16* Woh = (_Float16*)(ws + off); off += NW * 2;
  _Float16* Qh  = (_Float16*)(ws + off); off += NX * 2;
  _Float16* Kh  = (_Float16*)(ws + off); off += NX * 2;
  _Float16* Vt  = (_Float16*)(ws + off); off += NX * 2;
  _Float16* Ah  = (_Float16*)(ws + off); off += NX * 2;

  cvt_f32_f16_kernel<<<((int)NX + 255) / 256, 256, 0, stream>>>(x,  xh,  (int)NX);
  cvt_f32_f16_kernel<<<((int)NW + 255) / 256, 256, 0, stream>>>(Wq, Wqh, (int)NW);
  cvt_f32_f16_kernel<<<((int)NW + 255) / 256, 256, 0, stream>>>(Wk, Wkh, (int)NW);
  cvt_f32_f16_kernel<<<((int)NW + 255) / 256, 256, 0, stream>>>(Wv, Wvh, (int)NW);
  cvt_f32_f16_kernel<<<((int)NW + 255) / 256, 256, 0, stream>>>(Wo, Woh, (int)NW);

  // 3 matrices * 256 M-tiles * 12 N-strips = 9216 waves; 8 waves/block.
  qkv_gemm_kernel<<<1152, 256, 0, stream>>>(xh, Wqh, Wkh, Wvh, bq, bk, bv,
                                            Qh, Kh, Vt);
  // 12 heads * 32 blocks; each block = 8 waves sharing one head's K/V via TDM+LDS.
  attn_kernel<<<384, 256, 0, stream>>>(Qh, Kh, Vt, Ah);
  // 256 * 12 = 3072 waves.
  oproj_kernel<<<384, 256, 0, stream>>>(Ah, Woh, bo, out);
}
